// KStepRGCN_24747601559831
// MI455X (gfx1250) — matrix operand order repaired and checked by reading, not compiled, not run on hardware
//
#include <hip/hip_runtime.h>
#include <hip/hip_bf16.h>
#include <stddef.h>

// ---------------- problem constants ----------------
#define N_NODES 50000
#define N_EDGES 800000
#define CH      64          // IN_CH == OUT_CH
#define N_REL   8
#define N_BASES 4
#define NCAT    576         // 8 relations * 64 + 64 root columns
#define NTILES  36          // NCAT / 16
#define MTILES  3125        // N_NODES / 16

typedef __attribute__((ext_vector_type(16))) __bf16 v16bf;
typedef __attribute__((ext_vector_type(8)))  __bf16 v8bf;
typedef __attribute__((ext_vector_type(8)))  float  v8f;

// ---------------------------------------------------------------------------
// 1) Build the packed bf16 B-fragment buffer for the fused weight matrix
//    Wcat[k][n]:  n <  512 : sum_b att[r][b] * basis[b][k][o]   (r=n>>6, o=n&63)
//                 n >= 512 : root[k][n-512]
//    Packed layout: pB[(((t*2+s)*32 + lane)<<4) + j] holds
//      B element (K = s*32 + (lane>>4)*16 + j, N = t*16 + (lane&15))
//    i.e. contiguous k order (k = 16*(lane>>4) + j), the same order the A fragments use.
// ---------------------------------------------------------------------------
__global__ __launch_bounds__(256) void prep_weights_kernel(
    const float* __restrict__ basis,  // [4][64][64] for this layer
    const float* __restrict__ att,    // [8][4]      for this layer
    const float* __restrict__ root,   // [64][64]    for this layer
    __bf16* __restrict__ pB)          // [NTILES*2*32*16]
{
  int idx = blockIdx.x * blockDim.x + threadIdx.x;
  if (idx >= NTILES * 2 * 32 * 16) return;
  int j  = idx & 15;
  int l  = (idx >> 4) & 31;
  int ts = idx >> 9;
  int t  = ts >> 1;
  int s  = ts & 1;
  int kk = s * 32 + (l >> 4) * 16 + j;   // K index (input channel)
  int n  = t * 16 + (l & 15);            // fused output column
  float val;
  if (n < N_REL * CH) {
    int r = n >> 6, o = n & 63;
    val = 0.f;
#pragma unroll
    for (int b = 0; b < N_BASES; ++b)
      val = fmaf(att[r * N_BASES + b], basis[b * (CH * CH) + kk * CH + o], val);
  } else {
    val = root[kk * CH + (n - N_REL * CH)];
  }
  pB[idx] = (__bf16)val;
}

// ---------------------------------------------------------------------------
// 2) fp32 -> bf16 cast of the node features (layer-0 input only)
// ---------------------------------------------------------------------------
__global__ __launch_bounds__(256) void cast_kernel(
    const float* __restrict__ in, __bf16* __restrict__ out, int n)
{
  int i = blockIdx.x * blockDim.x + threadIdx.x;
  if (i < n) out[i] = (__bf16)in[i];
}

// ---------------------------------------------------------------------------
// 3) WMMA GEMM:  out[50000][576] = h_bf16[50000][64] @ Wcat[64][576]
//    grid = (3125 m-tiles, 9 column slots), one wave (32 threads) per block.
//    A-fragment (16-bit, 16x32): lane m (0-15) holds row m, K {0..15};
//    lane m+16 holds row m, K {16..31}  -> two 16-byte loads per step.
// ---------------------------------------------------------------------------
__global__ __launch_bounds__(32) void gemm_kernel(
    const __bf16* __restrict__ hb,   // [N_NODES][64] bf16
    const __bf16* __restrict__ pB,   // packed B fragments
    float* __restrict__ out)         // [N_NODES][576] f32
{
  const int m0   = blockIdx.x * 16;
  const int slot = blockIdx.y;           // 0..8, each covers 4 n-tiles
  const int lane = threadIdx.x;          // 0..31
  const int g    = lane >> 4;
  const int mrow = m0 + (lane & 15);

  v16bf a[2];
#pragma unroll
  for (int s = 0; s < 2; ++s) {
    const __bf16* pa = hb + (size_t)mrow * CH + s * 32 + g * 16;
    v8bf lo = *(const v8bf*)pa;          // K = s*32 + g*16 .. +7
    v8bf hi = *(const v8bf*)(pa + 8);    // K = s*32 + g*16 + 8 .. +7
#pragma unroll
    for (int j = 0; j < 8; ++j) { a[s][j] = lo[j]; a[s][8 + j] = hi[j]; }
  }

#pragma unroll
  for (int tt = 0; tt < 4; ++tt) {
    int t = slot * 4 + tt;
    v8f c = {};
#pragma unroll
    for (int s = 0; s < 2; ++s) {
      v16bf b = *(const v16bf*)(pB + (((size_t)(t * 2 + s) * 32 + lane) << 4));
      c = __builtin_amdgcn_wmma_f32_16x16x32_bf16(
              /*neg_a=*/false, a[s], /*neg_b=*/false, b,
              /*c_mod=*/(short)0, c, /*reuse_a=*/false, /*reuse_b=*/false);
    }
    // C layout: VGPR j -> row m0 + g*8 + j, col = t*16 + (lane&15)
    float* orow = out + (size_t)(m0 + g * 8) * NCAT + (size_t)t * 16 + (lane & 15);
#pragma unroll
    for (int j = 0; j < 8; ++j) orow[(size_t)j * NCAT] = c[j];
  }
}

// ---------------------------------------------------------------------------
// 4) Edge message + scatter-add.  One wave per edge (2 edges per wave with
//    prefetch of the second source row).  lane handles channels c and c+32.
//    msg[e][c] = sum_r attr[e][r] * xw[src[e]][r*64+c];  agg[dst[e]][c] += msg
// ---------------------------------------------------------------------------
__global__ __launch_bounds__(256) void edge_kernel(
    const int*   __restrict__ ei,    // [2][N_EDGES]
    const float* __restrict__ ea,    // [N_EDGES][8]
    const float* __restrict__ xw,    // [N_NODES][576]
    float*       __restrict__ agg)   // [N_NODES][64]
{
  unsigned wid  = blockIdx.x * (blockDim.x >> 5) + (threadIdx.x >> 5);
  int      lane = threadIdx.x & 31;
  unsigned e0   = wid * 2u;
  if (e0 >= N_EDGES) return;

  // prefetch second edge's gathered row (L2-resident; warms WGP$)
  {
    int src1 = ei[e0 + 1];
    __builtin_prefetch((const char*)(xw + (size_t)src1 * NCAT) + (size_t)lane * 64, 0, 0);
  }

#pragma unroll
  for (int q = 0; q < 2; ++q) {
    unsigned e = e0 + (unsigned)q;
    int src = ei[e];
    int dst = ei[N_EDGES + e];
    const float* row = xw + (size_t)src * NCAT;
    float acc0 = 0.f, acc1 = 0.f;
#pragma unroll
    for (int r = 0; r < N_REL; ++r) {
      float ar = ea[(size_t)e * N_REL + r];
      acc0 = fmaf(ar, row[r * CH + lane], acc0);
      acc1 = fmaf(ar, row[r * CH + 32 + lane], acc1);
    }
    atomicAdd(agg + (size_t)dst * CH + lane,      acc0);
    atomicAdd(agg + (size_t)dst * CH + 32 + lane, acc1);
  }
}

// ---------------------------------------------------------------------------
// 5) Finalize: h_next = agg + (h @ root) + bias  (root part lives in cols
//    512..575 of xwcat), optional ReLU, optional fp32 / bf16 outputs.
// ---------------------------------------------------------------------------
__global__ __launch_bounds__(256) void finalize_kernel(
    const float* __restrict__ agg,
    const float* __restrict__ xw,
    const float* __restrict__ bias,   // [64] for this layer
    float*  __restrict__ outf,        // fp32 output (may be null)
    __bf16* __restrict__ outbf,       // bf16 output for next layer (may be null)
    int do_relu)
{
  int i = blockIdx.x * blockDim.x + threadIdx.x;
  if (i >= N_NODES * CH) return;
  int n = i >> 6;
  int c = i & 63;
  float v = agg[i] + xw[(size_t)n * NCAT + N_REL * CH + c] + bias[c];
  if (do_relu) v = v > 0.f ? v : 0.f;
  if (outf)  outf[i]  = v;
  if (outbf) outbf[i] = (__bf16)v;
}

// ---------------------------------------------------------------------------
// host-side orchestration
// ---------------------------------------------------------------------------
extern "C" void kernel_launch(void* const* d_in, const int* in_sizes, int n_in,
                              void* d_out, int out_size, void* d_ws, size_t ws_size,
                              hipStream_t stream)
{
  const float* x          = (const float*)d_in[0];   // [50000][64]
  const int*   edge_index = (const int*)  d_in[1];   // [2][800000]
  const float* edge_attr  = (const float*)d_in[2];   // [800000][8]
  const float* basis      = (const float*)d_in[3];   // [2][4][64][64]
  const float* att        = (const float*)d_in[4];   // [2][8][4]
  const float* root       = (const float*)d_in[5];   // [2][64][64]
  const float* bias       = (const float*)d_in[6];   // [2][64]
  float*       out        = (float*)d_out;           // [50000][64]

  // workspace carve-up (all offsets 32B-aligned)
  char* ws = (char*)d_ws;
  float*  xwcat   = (float*) (ws + 0);                       // 115,200,000 B
  __bf16* h_bf16  = (__bf16*)(ws + 115200000);               //   6,400,000 B
  __bf16* packedB = (__bf16*)(ws + 121600000);               //      73,728 B
  float*  agg     = (float*) (ws + 121673728);               //  12,800,000 B
  (void)ws_size; (void)in_sizes; (void)n_in; (void)out_size;

  const int nElem = N_NODES * CH;                 // 3,200,000
  const size_t aggBytes = (size_t)nElem * sizeof(float);

  // layer 0 input cast
  cast_kernel<<<(nElem + 255) / 256, 256, 0, stream>>>(x, h_bf16, nElem);

  for (int k = 0; k < 2; ++k) {
    prep_weights_kernel<<<(NTILES * 2 * 32 * 16 + 255) / 256, 256, 0, stream>>>(
        basis + (size_t)k * N_BASES * CH * CH,
        att   + (size_t)k * N_REL * N_BASES,
        root  + (size_t)k * CH * CH,
        packedB);

    gemm_kernel<<<dim3(MTILES, 9), 32, 0, stream>>>(h_bf16, packedB, xwcat);

    hipMemsetAsync(agg, 0, aggBytes, stream);

    // 800000 edges, 2 per wave, 8 waves per block -> 50000 blocks
    edge_kernel<<<N_EDGES / 16, 256, 0, stream>>>(edge_index, edge_attr, xwcat, agg);

    finalize_kernel<<<(nElem + 255) / 256, 256, 0, stream>>>(
        agg, xwcat, bias + (size_t)k * CH,
        (k == 1) ? out : nullptr,
        (k == 0) ? h_bf16 : nullptr,
        /*do_relu=*/(k == 0) ? 1 : 0);
  }
}
